// MyMultiHeadAttention_84138409329121
// MI455X (gfx1250) — matrix operand order repaired and checked by reading, not compiled, not run on hardware
//
#include <hip/hip_runtime.h>
#include <hip/hip_bf16.h>

// ---------------------------------------------------------------------------
// MultiHeadAttention (channel-attention variant) for MI455X / gfx1250.
//   Stage 0: convert x, Wq, Wk, Wv f32 -> bf16
//   Stage 1: Q/K/V projections: TDM (tensor_load_to_lds) double-buffered LDS
//            tiles + v_wmma_f32_16x16x32_bf16.
//            Q,K stored head-transposed [head][d][S]; V stored [B*S][E]
//   Stage 2: scores[head] = Q_h (64xS) * K_h^T, K-split x8 into partials
//   Stage 2b: reduce + scale(1/8) + softmax -> bf16 attn [head][64][64]
//   Stage 3: out_h = attn * V_h, stored straight into final [B,S,E] f32 layout
// ---------------------------------------------------------------------------

typedef __attribute__((ext_vector_type(16))) __bf16          bf16x16;
typedef __attribute__((ext_vector_type(8)))  float           v8f;
typedef __attribute__((ext_vector_type(16))) unsigned short  u16x16;
typedef __attribute__((ext_vector_type(8)))  unsigned short  u16x8;
typedef __attribute__((ext_vector_type(4)))  unsigned int    u32x4;
typedef __attribute__((ext_vector_type(8)))  int             i32x8;
typedef __attribute__((ext_vector_type(4)))  int             i32x4;

#define EMB 1024
#define SEQ 4096
#define NHEAD_TOTAL 128   // B * H = 8 * 16

__device__ __forceinline__ unsigned short f32_to_bf16(float f) {
  unsigned int u = __float_as_uint(f);
  u += 0x7FFFu + ((u >> 16) & 1u);   // round-to-nearest-even
  return (unsigned short)(u >> 16);
}

// A fragment (16x32, bf16): lane m = lane&15, kh = lane>>4.
// element i -> K = k0 + kh*16 + i
__device__ __forceinline__ u16x16 load_frag_a(const unsigned short* base, int row0,
                                              int ld, int k0, int m_lane, int kh) {
  const unsigned short* p = base + (size_t)(row0 + m_lane) * ld + k0 + kh * 16;
  return *(const u16x16*)p;
}

// B fragment (32x16, bf16): lane n = lane&15, kh = lane>>4.
// element i -> K = k0 + kh*16 + i  (column-major B == row-major W[e][k])
__device__ __forceinline__ u16x16 load_frag_b(const unsigned short* base, int col0,
                                              int ld, int k0, int m_lane, int kh) {
  const unsigned short* p = base + (size_t)(col0 + m_lane) * ld + k0 + kh * 16;
  return *(const u16x16*)p;
}

__device__ __forceinline__ v8f wmma_bf16(u16x16 a, u16x16 b, v8f c) {
  union Cvt { u16x16 u; bf16x16 h; };
  Cvt ua, ub;
  ua.u = a; ub.u = b;
  return __builtin_amdgcn_wmma_f32_16x16x32_bf16(
      false, ua.h, false, ub.h, (short)0, c, false, false);
}

// ---------------------------------------------------------------------------
// TDM: 2D tile (tile_w elems per row, tile_h rows, element = 2 bytes) from
// global (row stride in elements) into LDS (packed, rows contiguous).
// D# group 0: count=1 | lds_addr | global_addr(57b) | type=2.
// D# group 1: data_size=1(2B) | tensor_dim0/1 | tile_dim0/1 | dim0_stride.
// Groups 2/3 (and the trailing group on this toolchain) zero: <=2D tensor.
// This toolchain's builtin takes 6 args:
//   (u32x4 g0, i32x8 g1, i32x4 g2, i32x4 g3, i32x8 gx, i32 cpol)
// ---------------------------------------------------------------------------
__device__ __forceinline__ void tdm_load_tile_2d(unsigned lds_off_bytes,
                                                 const void* gptr,
                                                 int tile_w, int tile_h,
                                                 int row_stride_elems) {
  unsigned long long ga = (unsigned long long)(size_t)gptr;
  u32x4 g0;
  g0[0] = 1u;                                             // count=1, user D#
  g0[1] = lds_off_bytes;                                  // lds_addr (bytes)
  g0[2] = (unsigned int)(ga & 0xFFFFFFFFull);             // global_addr[31:0]
  g0[3] = (unsigned int)((ga >> 32) & 0x1FFFFFFull)       // global_addr[56:32]
        | (2u << 30);                                     // type = 2 ("image")
  i32x8 g1;
  g1[0] = (1 << 16);                 // data_size=1 (2 bytes); no multicast
  g1[1] = (tile_w << 16);            // tensor_dim0[15:0]  (bits 63:48)
  g1[2] = (tile_h << 16);            // tensor_dim0 hi = 0; tensor_dim1[15:0]
  g1[3] = (tile_w << 16);            // tensor_dim1 hi = 0; tile_dim0
  g1[4] = tile_h;                    // tile_dim1; tile_dim2 = 0
  g1[5] = row_stride_elems;          // tensor_dim0_stride[31:0]
  g1[6] = 0;                         // stride hi / dim1_stride lo
  g1[7] = 0;                         // dim1_stride hi
  i32x4 gz4 = {0, 0, 0, 0};
  i32x8 gz8 = {0, 0, 0, 0, 0, 0, 0, 0};
  __builtin_amdgcn_tensor_load_to_lds(g0, g1, gz4, gz4, gz8, 0);
}

// Low 32 bits of the flat LDS address == byte offset within LDS aperture.
__device__ __forceinline__ unsigned lds_byte_offset(const void* p) {
  return (unsigned)(size_t)p;
}

// ---------------------------------------------------------------------------
// Stage 0: f32 -> bf16 conversion (4 elements per thread)
// ---------------------------------------------------------------------------
__global__ __launch_bounds__(256) void cvt_bf16_kernel(const float* __restrict__ in,
                                                       unsigned short* __restrict__ out,
                                                       int n4) {
  int i = blockIdx.x * blockDim.x + threadIdx.x;
  if (i >= n4) return;
  float4 f = ((const float4*)in)[i];
  unsigned int lo = (unsigned int)f32_to_bf16(f.x) | ((unsigned int)f32_to_bf16(f.y) << 16);
  unsigned int hi = (unsigned int)f32_to_bf16(f.z) | ((unsigned int)f32_to_bf16(f.w) << 16);
  unsigned int* o = (unsigned int*)out;
  o[2 * i] = lo;
  o[2 * i + 1] = hi;
}

// ---------------------------------------------------------------------------
// Stage 1: P = x @ W^T + b.  Block = 8 waves (4m x 2n), wave tile 32x32.
// TDM double-buffered LDS staging of A (128x32) and B (64x32) tiles.
// mode 0: store head-transposed bf16 [batch*1024 + e][SEQ] (Q, K)
// mode 1: store natural bf16 [row][EMB]                    (V)
// ---------------------------------------------------------------------------
__global__ __launch_bounds__(256) void qkv_proj_kernel(
    const unsigned short* __restrict__ X,    // [32768][1024] bf16
    const unsigned short* __restrict__ Wb,   // [1024][1024]  bf16
    const float* __restrict__ bias,          // [1024] f32
    unsigned short* __restrict__ outT,       // [B*1024][SEQ] bf16 (mode 0)
    unsigned short* __restrict__ outN,       // [32768][1024] bf16 (mode 1)
    int mode) {
  __shared__ unsigned short A_lds[2][128][32];   // 2 x 8 KB
  __shared__ unsigned short B_lds[2][64][32];    // 2 x 4 KB

  const int lane = threadIdx.x & 31;
  const int wave = threadIdx.x >> 5;
  const int m_lane = lane & 15;
  const int kh = lane >> 4;
  const int rowblk = blockIdx.x * 128;           // block M origin
  const int colblk = blockIdx.y * 64;            // block N origin
  const int wr = (wave >> 1) * 32;               // wave M offset in block
  const int wc = (wave & 1) * 32;                // wave N offset in block

  v8f acc[2][2];
#pragma unroll
  for (int mi = 0; mi < 2; ++mi)
#pragma unroll
    for (int ni = 0; ni < 2; ++ni)
#pragma unroll
      for (int r = 0; r < 8; ++r) acc[mi][ni][r] = 0.0f;

  const int NIT = EMB / 32;   // 32 k-steps

  // Prologue: wave 0 issues TDM for k-step 0 into buffer 0.
  if (wave == 0) {
    tdm_load_tile_2d(lds_byte_offset(&A_lds[0][0][0]),
                     X + (size_t)rowblk * EMB, 32, 128, EMB);
    tdm_load_tile_2d(lds_byte_offset(&B_lds[0][0][0]),
                     Wb + (size_t)colblk * EMB, 32, 64, EMB);
  }

  for (int it = 0; it < NIT; ++it) {
    const int buf = it & 1;
    if (wave == 0) {
      if (it + 1 < NIT) {
        const int k0n = (it + 1) * 32;
        tdm_load_tile_2d(lds_byte_offset(&A_lds[buf ^ 1][0][0]),
                         X + (size_t)rowblk * EMB + k0n, 32, 128, EMB);
        tdm_load_tile_2d(lds_byte_offset(&B_lds[buf ^ 1][0][0]),
                         Wb + (size_t)colblk * EMB + k0n, 32, 64, EMB);
        __builtin_amdgcn_s_wait_tensorcnt(2);  // current buffer complete
      } else {
        __builtin_amdgcn_s_wait_tensorcnt(0);
      }
    }
    __syncthreads();   // publish buf to all waves

    u16x16 a0 = load_frag_a(&A_lds[buf][0][0], wr, 32, 0, m_lane, kh);
    u16x16 a1 = load_frag_a(&A_lds[buf][0][0], wr + 16, 32, 0, m_lane, kh);
    u16x16 b0 = load_frag_b(&B_lds[buf][0][0], wc, 32, 0, m_lane, kh);
    u16x16 b1 = load_frag_b(&B_lds[buf][0][0], wc + 16, 32, 0, m_lane, kh);
    acc[0][0] = wmma_bf16(a0, b0, acc[0][0]);
    acc[0][1] = wmma_bf16(a0, b1, acc[0][1]);
    acc[1][0] = wmma_bf16(a1, b0, acc[1][0]);
    acc[1][1] = wmma_bf16(a1, b1, acc[1][1]);

    __syncthreads();   // all reads of buf done before TDM may overwrite it
  }

  const int rowbase = rowblk + wr;
  const int colbase = colblk + wc;
  const int batch = rowbase / SEQ;           // block tile never straddles a batch
  const int s_in_batch = rowbase % SEQ;

#pragma unroll
  for (int ni = 0; ni < 2; ++ni) {
    const int e = colbase + ni * 16 + m_lane;      // output feature (per-lane)
    const float bv = bias[e];
#pragma unroll
    for (int mi = 0; mi < 2; ++mi) {
      v8f c = acc[mi][ni];
      if (mode == 0) {
        const int s0 = s_in_batch + mi * 16 + kh * 8;
        u16x8 pk;
#pragma unroll
        for (int r = 0; r < 8; ++r) pk[r] = f32_to_bf16(c[r] + bv);
        *(u16x8*)(outT + ((size_t)(batch * EMB + e)) * SEQ + s0) = pk;
      } else {
#pragma unroll
        for (int r = 0; r < 8; ++r) {
          const int row = rowbase + mi * 16 + kh * 8 + r;
          outN[(size_t)row * EMB + e] = f32_to_bf16(c[r] + bv);
        }
      }
    }
  }
}

// ---------------------------------------------------------------------------
// Stage 2: partial scores. grid = (128 heads, 8 K-splits), block = 2 waves.
// ---------------------------------------------------------------------------
__global__ __launch_bounds__(64) void scores_kernel(
    const unsigned short* __restrict__ Qt,   // [128*64][SEQ] bf16
    const unsigned short* __restrict__ Kt,   // [128*64][SEQ] bf16
    float* __restrict__ part) {              // [8][128][64][64] f32
  const int head = blockIdx.x;
  const int ksp = blockIdx.y;
  const int lane = threadIdx.x & 31;
  const int wave = threadIdx.x >> 5;
  const int m_lane = lane & 15;
  const int kh = lane >> 4;

  const unsigned short* Q = Qt + (size_t)head * 64 * SEQ;
  const unsigned short* K = Kt + (size_t)head * 64 * SEQ;
  const int d0 = wave * 32;

  v8f acc[2][4];
#pragma unroll
  for (int mi = 0; mi < 2; ++mi)
#pragma unroll
    for (int ni = 0; ni < 4; ++ni)
#pragma unroll
      for (int r = 0; r < 8; ++r) acc[mi][ni][r] = 0.0f;

  const int kbeg = ksp * 512;
  for (int k0 = kbeg; k0 < kbeg + 512; k0 += 32) {
    u16x16 a0 = load_frag_a(Q, d0, SEQ, k0, m_lane, kh);
    u16x16 a1 = load_frag_a(Q, d0 + 16, SEQ, k0, m_lane, kh);
#pragma unroll
    for (int ni = 0; ni < 4; ++ni) {
      u16x16 b = load_frag_b(K, ni * 16, SEQ, k0, m_lane, kh);
      acc[0][ni] = wmma_bf16(a0, b, acc[0][ni]);
      acc[1][ni] = wmma_bf16(a1, b, acc[1][ni]);
    }
  }

  float* out = part + (size_t)ksp * (NHEAD_TOTAL * 64 * 64) + (size_t)head * 4096;
#pragma unroll
  for (int mi = 0; mi < 2; ++mi)
#pragma unroll
    for (int ni = 0; ni < 4; ++ni)
#pragma unroll
      for (int r = 0; r < 8; ++r) {
        const int m = d0 + mi * 16 + kh * 8 + r;
        const int n = ni * 16 + m_lane;
        out[m * 64 + n] = acc[mi][ni][r];
      }
}

// ---------------------------------------------------------------------------
// Stage 2b: reduce 8 partials, scale by 0.125, softmax over 64, emit bf16.
// ---------------------------------------------------------------------------
__global__ __launch_bounds__(128) void softmax_kernel(
    const float* __restrict__ part, unsigned short* __restrict__ attnb) {
  const int row = blockIdx.x * 4 + (threadIdx.x >> 5);   // 8192 rows
  const int lane = threadIdx.x & 31;
  const size_t base = (size_t)row * 64 + lane * 2;

  float a = 0.0f, b = 0.0f;
#pragma unroll
  for (int p = 0; p < 8; ++p) {
    a += part[(size_t)p * (NHEAD_TOTAL * 64 * 64) + base];
    b += part[(size_t)p * (NHEAD_TOTAL * 64 * 64) + base + 1];
  }
  a *= 0.125f;
  b *= 0.125f;

  float m = fmaxf(a, b);
#pragma unroll
  for (int off = 16; off > 0; off >>= 1) m = fmaxf(m, __shfl_xor(m, off, 32));
  float ea = __expf(a - m), eb = __expf(b - m);
  float s = ea + eb;
#pragma unroll
  for (int off = 16; off > 0; off >>= 1) s += __shfl_xor(s, off, 32);
  const float inv = 1.0f / s;

  attnb[base] = f32_to_bf16(ea * inv);
  attnb[base + 1] = f32_to_bf16(eb * inv);
}

// ---------------------------------------------------------------------------
// Stage 3: out_h = attn (64x64) * V_h (64xS) -> final [B,S,E] f32 layout.
// ---------------------------------------------------------------------------
__global__ __launch_bounds__(256) void av_kernel(
    const unsigned short* __restrict__ attnb,  // [128][64][64] bf16
    const unsigned short* __restrict__ Vn,     // [32768][1024] bf16
    float* __restrict__ out) {                 // [B][S][E] f32
  const int lane = threadIdx.x & 31;
  const int wave = threadIdx.x >> 5;
  const int m_lane = lane & 15;
  const int kh = lane >> 4;

  const int head = blockIdx.x >> 5;                 // 32 blocks per head
  const int widx = (blockIdx.x & 31) * 8 + wave;    // 0..255 within head
  const int dt = widx & 3;                          // d tile (16 rows)
  const int sg = widx >> 2;                         // s group (64 cols)
  const int batch = head >> 4;
  const int h = head & 15;

  const unsigned short* A = attnb + (size_t)head * 4096;  // 64x64 row-major

  v8f acc[4];
#pragma unroll
  for (int ni = 0; ni < 4; ++ni)
#pragma unroll
    for (int r = 0; r < 8; ++r) acc[ni][r] = 0.0f;

#pragma unroll
  for (int k0 = 0; k0 < 64; k0 += 32) {
    u16x16 a = load_frag_a(A, dt * 16, 64, k0, m_lane, kh);
#pragma unroll
    for (int ni = 0; ni < 4; ++ni) {
      const int s = sg * 64 + ni * 16 + m_lane;
      const unsigned short* p =
          Vn + ((size_t)(batch * SEQ + s)) * EMB + h * 64 + k0 + kh * 16;
      u16x16 b = *(const u16x16*)p;
      acc[ni] = wmma_bf16(a, b, acc[ni]);
    }
  }

#pragma unroll
  for (int ni = 0; ni < 4; ++ni) {
    const int s = sg * 64 + ni * 16 + m_lane;
    float* o = out + ((size_t)(batch * SEQ + s)) * EMB + h * 64 + dt * 16 + kh * 8;
    *(v8f*)o = acc[ni];   // 8 consecutive f32 (32B aligned)
  }
}

// ---------------------------------------------------------------------------
// Host-side launch
// ---------------------------------------------------------------------------
extern "C" void kernel_launch(void* const* d_in, const int* in_sizes, int n_in,
                              void* d_out, int out_size, void* d_ws, size_t ws_size,
                              hipStream_t stream) {
  const float* x  = (const float*)d_in[0];
  const float* Wq = (const float*)d_in[1];
  const float* bq = (const float*)d_in[2];
  const float* Wk = (const float*)d_in[3];
  const float* bk = (const float*)d_in[4];
  const float* Wv = (const float*)d_in[5];
  const float* bv = (const float*)d_in[6];
  float* out = (float*)d_out;

  char* ws = (char*)d_ws;
  const size_t SZ_XB = (size_t)32768 * 1024 * 2;    // 64 MB
  const size_t SZ_W  = (size_t)1024 * 1024 * 2;     // 2 MB each
  const size_t SZ_T  = (size_t)128 * 64 * 4096 * 2; // 64 MB each (Qt/Kt/Vn)
  unsigned short* xb    = (unsigned short*)(ws);
  unsigned short* Wqb   = (unsigned short*)(ws + SZ_XB);
  unsigned short* Wkb   = (unsigned short*)(ws + SZ_XB + SZ_W);
  unsigned short* Wvb   = (unsigned short*)(ws + SZ_XB + 2 * SZ_W);
  unsigned short* Qt    = (unsigned short*)(ws + SZ_XB + 3 * SZ_W);
  unsigned short* Kt    = (unsigned short*)(ws + SZ_XB + 3 * SZ_W + SZ_T);
  unsigned short* Vn    = (unsigned short*)(ws + SZ_XB + 3 * SZ_W + 2 * SZ_T);
  float*          part  = (float*)(ws + SZ_XB + 3 * SZ_W + 3 * SZ_T);
  unsigned short* attnb = (unsigned short*)(ws + SZ_XB + 3 * SZ_W + 3 * SZ_T +
                                            (size_t)8 * 128 * 64 * 64 * 4);

  // Stage 0: conversions
  cvt_bf16_kernel<<<32768, 256, 0, stream>>>(x, xb, 33554432 / 4);
  cvt_bf16_kernel<<<1024, 256, 0, stream>>>(Wq, Wqb, 1048576 / 4);
  cvt_bf16_kernel<<<1024, 256, 0, stream>>>(Wk, Wkb, 1048576 / 4);
  cvt_bf16_kernel<<<1024, 256, 0, stream>>>(Wv, Wvb, 1048576 / 4);

  // Stage 1: projections (M=32768, N=1024, K=1024), TDM-staged LDS tiles
  dim3 g1(256, 16);
  qkv_proj_kernel<<<g1, 256, 0, stream>>>(xb, Wqb, bq, Qt, nullptr, 0);
  qkv_proj_kernel<<<g1, 256, 0, stream>>>(xb, Wkb, bk, Kt, nullptr, 0);
  qkv_proj_kernel<<<g1, 256, 0, stream>>>(xb, Wvb, bv, nullptr, Vn, 1);

  // Stage 2: Q*K^T partials (K split by 8, deterministic)
  scores_kernel<<<dim3(128, 8), 64, 0, stream>>>(Qt, Kt, part);

  // Stage 2b: reduce + scale + softmax -> bf16 attn
  softmax_kernel<<<2048, 128, 0, stream>>>(part, attnb);

  // Stage 3: attn * V -> final [B,S,E] f32
  av_kernel<<<4096, 256, 0, stream>>>(attnb, Vn, out);

  (void)in_sizes; (void)n_in; (void)out_size; (void)ws_size;
}
